// TransformerBlock_28209345200691
// MI455X (gfx1250) — compile-verified, matrix operand order checked by reading, not run on hardware
//
#include <hip/hip_runtime.h>
#include <math.h>

typedef __bf16 bf16;
typedef __attribute__((ext_vector_type(16))) __bf16 v16bf;
typedef __attribute__((ext_vector_type(8)))  __bf16 v8bf;
typedef __attribute__((ext_vector_type(8)))  float  v8f;

constexpr int BB = 2, SS = 2048, DD = 1024, HH = 16, KVH = 8, HDim = 128, FFD = 3072;
constexpr int MR = BB * SS; // 4096 token rows

// ---------- scalar bf16 helpers (RNE) ----------
__device__ __forceinline__ bf16 f2bf(float f) {
  unsigned u = __builtin_bit_cast(unsigned, f);
  u += 0x7FFFu + ((u >> 16) & 1u);
  unsigned short s = (unsigned short)(u >> 16);
  return __builtin_bit_cast(bf16, s);
}
__device__ __forceinline__ float bf2f(bf16 b) {
  unsigned short s = __builtin_bit_cast(unsigned short, b);
  unsigned u = (unsigned)s << 16;
  return __builtin_bit_cast(float, u);
}
__device__ __forceinline__ unsigned f2bf_pk(float lo, float hi) {
  unsigned ul = __builtin_bit_cast(unsigned, lo);
  ul += 0x7FFFu + ((ul >> 16) & 1u);
  unsigned uh = __builtin_bit_cast(unsigned, hi);
  uh += 0x7FFFu + ((uh >> 16) & 1u);
  return (ul >> 16) | (uh & 0xFFFF0000u);
}
__device__ __forceinline__ v8f zero8() {
  v8f z;
#pragma unroll
  for (int i = 0; i < 8; ++i) z[i] = 0.f;
  return z;
}

// ---------- WMMA fragment helpers ----------
// 16-bit A/B fragment, 16x32 (MxK) layout per ISA 7.12.2:
//   lanes 0-15 : row = lane,     elems K=[0..7] and K=[16..23]
//   lanes 16-31: row = lane-16,  elems K=[8..15] and K=[24..31]
__device__ __forceinline__ v16bf load_frag(const bf16* smem, int row0, int stride, int koff) {
  int lane = threadIdx.x & 31;
  const bf16* p = smem + (size_t)(row0 + (lane & 15)) * stride + koff + ((lane >> 4) << 3);
  v8bf lo = *(const v8bf*)(p);
  v8bf hi = *(const v8bf*)(p + 16);
  v16bf f;
#pragma unroll
  for (int i = 0; i < 8; ++i) { f[i] = lo[i]; f[i + 8] = hi[i]; }
  return f;
}

__device__ __forceinline__ v8f wmma_bf16(v16bf a, v16bf b, v8f c) {
  return __builtin_amdgcn_wmma_f32_16x16x32_bf16(false, a, false, b, (short)0, c, false, false);
}

// ---------- weight convert + transpose: W[K,N] f32 -> Wt[N,K] bf16 ----------
__global__ __launch_bounds__(256) void convtrans_kernel(const float* __restrict__ W,
                                                        bf16* __restrict__ Wt, int K, int N) {
  size_t i = (size_t)blockIdx.x * 256 + threadIdx.x;
  if (i >= (size_t)K * N) return;
  int n = (int)(i % N);
  int k = (int)(i / N);
  Wt[(size_t)n * K + k] = f2bf(W[i]);
}

// ---------- rmsnorm over D=1024, fp32 in -> bf16 out; 256 thr, 4 elems/thr ----------
__global__ __launch_bounds__(256) void rmsnorm_kernel(const float* __restrict__ x,
                                                      const float* __restrict__ scale,
                                                      bf16* __restrict__ out, int D) {
  const size_t row = blockIdx.x;
  const int tid = threadIdx.x, lane = tid & 31, wid = tid >> 5;
  const float4 v = *(const float4*)(x + row * D + tid * 4);
  float ss = v.x * v.x + v.y * v.y + v.z * v.z + v.w * v.w;
#pragma unroll
  for (int m = 16; m >= 1; m >>= 1) ss += __shfl_xor(ss, m, 32);
  __shared__ float red[8];
  if (lane == 0) red[wid] = ss;
  __syncthreads();
  float tot = 0.f;
#pragma unroll
  for (int i = 0; i < 8; ++i) tot += red[i];
  const float r = rsqrtf(tot / (float)D + 1e-6f);
  const float4 sc = *(const float4*)(scale + tid * 4);
  unsigned2_tag: ;
  unsigned p0 = f2bf_pk(v.x * r * sc.x, v.y * r * sc.y);
  unsigned p1 = f2bf_pk(v.z * r * sc.z, v.w * r * sc.w);
  uint2 pk; pk.x = p0; pk.y = p1;
  *(uint2*)(out + row * D + tid * 4) = pk;
}

// ---------- bf16 WMMA GEMM: C[M,N] = A[M,K] @ B[K,N], B given transposed (Bt[N,K]) ----
// Block tile 128x256, 8 waves in 2(M)x4(N) grid, wave tile 64x64 (16 accumulators).
// Register double-buffering of the next K-slab (BK=32).
// EPI==0 : out bf16 = C            EPI==1 : out f32 = resid + C
template <int EPI>
__global__ __launch_bounds__(256) void gemm_kernel(const bf16* __restrict__ A,
                                                   const bf16* __restrict__ Bt,
                                                   int M, int N, int K,
                                                   bf16* __restrict__ outb,
                                                   float* __restrict__ outf,
                                                   const float* __restrict__ resid) {
  __shared__ bf16 As[128 * 32];   // 8KB
  __shared__ bf16 Bs[256 * 32];   // 16KB
  const int tid = threadIdx.x;
  const int lane = tid & 31, wid = tid >> 5;
  const int wm = wid & 1, wn = wid >> 1; // 2 x 4 wave grid
  const size_t bm = (size_t)blockIdx.x * 128, bn = (size_t)blockIdx.y * 256;

  v8f acc[4][4];
#pragma unroll
  for (int mi = 0; mi < 4; ++mi)
#pragma unroll
    for (int ni = 0; ni < 4; ++ni) acc[mi][ni] = zero8();

  // per-thread staging: A = 2 chunks of 8 bf16, B = 4 chunks of 8 bf16
  v8bf areg[2], breg[4];
  auto load_stage = [&](int k0) {
#pragma unroll
    for (int i = 0; i < 2; ++i) {
      int c = tid * 2 + i;                 // 512 A-chunks
      int r = c >> 2, cc = (c & 3) << 3;
      areg[i] = *(const v8bf*)(A + (bm + r) * K + k0 + cc);
    }
#pragma unroll
    for (int i = 0; i < 4; ++i) {
      int c = tid * 4 + i;                 // 1024 B-chunks
      int r = c >> 2, cc = (c & 3) << 3;
      breg[i] = *(const v8bf*)(Bt + (bn + r) * K + k0 + cc);
    }
  };

  load_stage(0);
  for (int k0 = 0; k0 < K; k0 += 32) {
    __syncthreads();                       // previous compute done
#pragma unroll
    for (int i = 0; i < 2; ++i) {
      int c = tid * 2 + i;
      int r = c >> 2, cc = (c & 3) << 3;
      *(v8bf*)(As + r * 32 + cc) = areg[i];
    }
#pragma unroll
    for (int i = 0; i < 4; ++i) {
      int c = tid * 4 + i;
      int r = c >> 2, cc = (c & 3) << 3;
      *(v8bf*)(Bs + r * 32 + cc) = breg[i];
    }
    __syncthreads();
    if (k0 + 32 < K) load_stage(k0 + 32);  // global latency hides under WMMAs

    v16bf af[4];
#pragma unroll
    for (int mi = 0; mi < 4; ++mi) af[mi] = load_frag(As, wm * 64 + mi * 16, 32, 0);
#pragma unroll
    for (int ni = 0; ni < 4; ++ni) {
      v16bf bfr = load_frag(Bs, wn * 64 + ni * 16, 32, 0);
#pragma unroll
      for (int mi = 0; mi < 4; ++mi) acc[mi][ni] = wmma_bf16(af[mi], bfr, acc[mi][ni]);
    }
  }

  const int hi = lane >> 4, cl = lane & 15;
#pragma unroll
  for (int mi = 0; mi < 4; ++mi)
#pragma unroll
    for (int ni = 0; ni < 4; ++ni)
#pragma unroll
      for (int i = 0; i < 8; ++i) {
        size_t row = bm + wm * 64 + mi * 16 + i + hi * 8;
        size_t col = bn + wn * 64 + ni * 16 + cl;
        float v = acc[mi][ni][i];
        if (EPI == 0) outb[row * N + col] = f2bf(v);
        else          outf[row * N + col] = resid[row * N + col] + v;
      }
}

// ---------- per-(b,s,head) rmsnorm + RoPE; one wave per row of 128 ----------
__global__ __launch_bounds__(128) void rope_norm_kernel(const bf16* __restrict__ src,
                                                        bf16* __restrict__ dst,
                                                        const float* __restrict__ nscale,
                                                        const float* __restrict__ cosb,
                                                        const float* __restrict__ sinb,
                                                        int NH, float outScale) {
  const int w = threadIdx.x >> 5, lane = threadIdx.x & 31;
  const int rowid = blockIdx.x * 4 + w; // [0, BB*SS*NH)
  const int h = rowid % NH;
  const int bs = rowid / NH;
  const int s = bs % SS;
  const int b = bs / SS;
  const bf16* p = src + ((size_t)bs * NH + h) * HDim;
  float e[4];
#pragma unroll
  for (int j = 0; j < 4; ++j) e[j] = bf2f(p[lane + 32 * j]);
  float ss = e[0] * e[0] + e[1] * e[1] + e[2] * e[2] + e[3] * e[3];
#pragma unroll
  for (int m = 16; m >= 1; m >>= 1) ss += __shfl_xor(ss, m, 32);
  float r = rsqrtf(ss * (1.f / HDim) + 1e-6f);
  float n[4], c[4], sn[4];
#pragma unroll
  for (int j = 0; j < 4; ++j) {
    int d = lane + 32 * j;
    n[j]  = e[j] * r * nscale[d];
    c[j]  = cosb[(size_t)s * HDim + d];
    sn[j] = sinb[(size_t)s * HDim + d];
  }
  // rotate-half: out[d] = x[d]*cos[d] - x[d+64]*sin[d]  (d<64)
  //              out[d] = x[d]*cos[d] + x[d-64]*sin[d]  (d>=64)
  float o0 = n[0] * c[0] - n[2] * sn[0];
  float o1 = n[1] * c[1] - n[3] * sn[1];
  float o2 = n[2] * c[2] + n[0] * sn[2];
  float o3 = n[3] * c[3] + n[1] * sn[3];
  bf16* q = dst + (((size_t)b * NH + h) * SS + s) * HDim;
  q[lane]      = f2bf(o0 * outScale);
  q[lane + 32] = f2bf(o1 * outScale);
  q[lane + 64] = f2bf(o2 * outScale);
  q[lane + 96] = f2bf(o3 * outScale);
}

// ---------- V: [B,S,KV*HD] bf16 -> V^T [B,KV,HD,S] bf16 ----------
__global__ __launch_bounds__(256) void vtrans_kernel(const bf16* __restrict__ src,
                                                     bf16* __restrict__ dst) {
  size_t idx = (size_t)blockIdx.x * 256 + threadIdx.x;
  int d = (int)(idx % HDim);
  size_t t = idx / HDim;
  int h = (int)(t % KVH);
  size_t t2 = t / KVH;
  int s = (int)(t2 % SS);
  int b = (int)(t2 / SS);
  dst[(((size_t)b * KVH + h) * HDim + d) * SS + s] = src[idx];
}

// ---------- flash attention: Q[B,H,S,HD] (pre-scaled), K[B,KV,S,HD], Vt[B,KV,HD,S] ----
__global__ __launch_bounds__(128) void attn_kernel(const bf16* __restrict__ qr,
                                                   const bf16* __restrict__ kr,
                                                   const bf16* __restrict__ vt,
                                                   bf16* __restrict__ ctx) {
  __shared__ bf16 Qs[64 * HDim];   // 16KB
  __shared__ bf16 Ks[64 * HDim];   // 16KB
  __shared__ bf16 Vts[HDim * 64];  // 16KB  (rows=d, cols=k-local)
  __shared__ bf16 Ps[64 * 64];     // 8KB

  const int bh = blockIdx.x;   // b*HH + h
  const int qb = blockIdx.y;   // q block of 64
  const int b = bh / HH, h = bh % HH;
  const int hk = h / (HH / KVH);
  const int tid = threadIdx.x, w = tid >> 5, lane = tid & 31;
  const int hi = lane >> 4, cl = lane & 15;

  const bf16* qbase = qr + (((size_t)b * HH + h) * SS + (size_t)qb * 64) * HDim;
  for (int i = tid; i < 64 * 16; i += 128) {
    int r = i >> 4, c = (i & 15) << 3;
    *(v8bf*)(Qs + r * HDim + c) = *(const v8bf*)(qbase + (size_t)r * HDim + c);
  }

  v8f oacc[8];
#pragma unroll
  for (int di = 0; di < 8; ++di) oacc[di] = zero8();
  float mrow[8], lrow[8];
#pragma unroll
  for (int i = 0; i < 8; ++i) { mrow[i] = -1e30f; lrow[i] = 0.f; }

  const bf16* kbase = kr + ((size_t)b * KVH + hk) * SS * HDim;
  const bf16* vbase = vt + ((size_t)b * KVH + hk) * (size_t)HDim * SS;

  for (int kb = 0; kb <= qb; ++kb) {
    __syncthreads();
    for (int i = tid; i < 64 * 16; i += 128) {
      int r = i >> 4, c = (i & 15) << 3;
      *(v8bf*)(Ks + r * HDim + c) = *(const v8bf*)(kbase + (size_t)(kb * 64 + r) * HDim + c);
    }
    for (int i = tid; i < 128 * 8; i += 128) {
      int r = i >> 3, c = (i & 7) << 3;
      *(v8bf*)(Vts + r * 64 + c) = *(const v8bf*)(vbase + (size_t)r * SS + kb * 64 + c);
    }
    __syncthreads();

    // S = Q @ K^T  : wave w owns q rows [w*16, w*16+16), all 64 k cols
    v8f sacc[4];
#pragma unroll
    for (int ni = 0; ni < 4; ++ni) sacc[ni] = zero8();
#pragma unroll
    for (int ks = 0; ks < 4; ++ks) {
      v16bf aq = load_frag(Qs, w * 16, HDim, ks * 32);
#pragma unroll
      for (int ni = 0; ni < 4; ++ni) {
        v16bf bk = load_frag(Ks, ni * 16, HDim, ks * 32);
        sacc[ni] = wmma_bf16(aq, bk, sacc[ni]);
      }
    }

    if (kb == qb) { // causal mask on the diagonal block (blocks are aligned)
#pragma unroll
      for (int ni = 0; ni < 4; ++ni)
#pragma unroll
        for (int i = 0; i < 8; ++i)
          if (ni * 16 + cl > w * 16 + i + 8 * hi) sacc[ni][i] = -1e30f;
    }

    // online softmax; C-layout: row (i + 8*hi) lives in the lane's 16-lane half
#pragma unroll
    for (int i = 0; i < 8; ++i) {
      float mx = fmaxf(fmaxf(sacc[0][i], sacc[1][i]), fmaxf(sacc[2][i], sacc[3][i]));
#pragma unroll
      for (int msk = 8; msk >= 1; msk >>= 1) mx = fmaxf(mx, __shfl_xor(mx, msk, 32));
      float mn = fmaxf(mrow[i], mx);
      float scl = __expf(mrow[i] - mn);
      mrow[i] = mn;
      float rs = 0.f;
#pragma unroll
      for (int ni = 0; ni < 4; ++ni) {
        float pv = __expf(sacc[ni][i] - mn);
        sacc[ni][i] = pv;
        rs += pv;
      }
#pragma unroll
      for (int msk = 8; msk >= 1; msk >>= 1) rs += __shfl_xor(rs, msk, 32);
      lrow[i] = lrow[i] * scl + rs;
#pragma unroll
      for (int di = 0; di < 8; ++di) oacc[di][i] *= scl;
    }

    // P -> LDS (bf16), then O += P @ V  (same-wave LDS ops are in-order)
#pragma unroll
    for (int ni = 0; ni < 4; ++ni)
#pragma unroll
      for (int i = 0; i < 8; ++i)
        Ps[(size_t)(w * 16 + i + 8 * hi) * 64 + ni * 16 + cl] = f2bf(sacc[ni][i]);
#pragma unroll
    for (int ks = 0; ks < 2; ++ks) {
      v16bf pa = load_frag(Ps, w * 16, 64, ks * 32);
#pragma unroll
      for (int di = 0; di < 8; ++di) {
        v16bf bv = load_frag(Vts, di * 16, 64, ks * 32);
        oacc[di] = wmma_bf16(pa, bv, oacc[di]);
      }
    }
  }

  // write ctx[b, q, h*HD + d]
#pragma unroll
  for (int di = 0; di < 8; ++di)
#pragma unroll
    for (int i = 0; i < 8; ++i) {
      size_t r = (size_t)qb * 64 + w * 16 + i + 8 * hi;
      int d = di * 16 + cl;
      float v = oacc[di][i] / lrow[i];
      ctx[((size_t)b * SS + r) * (HH * HDim) + (size_t)h * HDim + d] = f2bf(v);
    }
}

// ---------- silu(a) * b ----------
__global__ __launch_bounds__(256) void silu_mul_kernel(const bf16* __restrict__ a,
                                                       const bf16* __restrict__ b,
                                                       bf16* __restrict__ o) {
  size_t i = (size_t)blockIdx.x * 256 + threadIdx.x;
  float av = bf2f(a[i]), bv = bf2f(b[i]);
  float s = av / (1.f + __expf(-av));
  o[i] = f2bf(s * bv);
}

// =====================================================================================
extern "C" void kernel_launch(void* const* d_in, const int* in_sizes, int n_in,
                              void* d_out, int out_size, void* d_ws, size_t ws_size,
                              hipStream_t stream) {
  const float* x    = (const float*)d_in[0];
  // d_in[1] = mask (unused: causal mask applied analytically)
  const float* cosb = (const float*)d_in[2];
  const float* sinb = (const float*)d_in[3];
  const float* wq   = (const float*)d_in[4];
  const float* wk   = (const float*)d_in[5];
  const float* wv   = (const float*)d_in[6];
  const float* wo   = (const float*)d_in[7];
  const float* fc1  = (const float*)d_in[8];
  const float* fc2  = (const float*)d_in[9];
  const float* fc3  = (const float*)d_in[10];
  const float* n1s  = (const float*)d_in[11];
  const float* n2s  = (const float*)d_in[12];
  const float* qns  = (const float*)d_in[13];
  const float* kns  = (const float*)d_in[14];
  float* out = (float*)d_out;
  (void)in_sizes; (void)n_in; (void)out_size; (void)ws_size;

  char* ws = (char*)d_ws;
  size_t off = 0;
  auto take = [&](size_t bytes) -> char* {
    char* r = ws + off;
    off += (bytes + 255) & ~(size_t)255;
    return r;
  };
  bf16* wqT  = (bf16*)take((size_t)2048 * 1024 * 2);
  bf16* wkT  = (bf16*)take((size_t)1024 * 1024 * 2);
  bf16* wvT  = (bf16*)take((size_t)1024 * 1024 * 2);
  bf16* woT  = (bf16*)take((size_t)1024 * 2048 * 2);
  bf16* fc1T = (bf16*)take((size_t)3072 * 1024 * 2);
  bf16* fc2T = (bf16*)take((size_t)3072 * 1024 * 2);
  bf16* fc3T = (bf16*)take((size_t)1024 * 3072 * 2);
  bf16* hbuf = (bf16*)take((size_t)MR * DD * 2);
  bf16* qbuf = (bf16*)take((size_t)MR * HH * HDim * 2);   // 16MB
  bf16* kbuf = (bf16*)take((size_t)MR * KVH * HDim * 2);  // 8MB
  bf16* vbuf = (bf16*)take((size_t)MR * KVH * HDim * 2);  // 8MB
  bf16* qr   = (bf16*)take((size_t)MR * HH * HDim * 2);   // 16MB
  bf16* kr   = (bf16*)take((size_t)MR * KVH * HDim * 2);  // 8MB
  bf16* vtb  = (bf16*)take((size_t)MR * KVH * HDim * 2);  // 8MB
  float* x1  = (float*)take((size_t)MR * DD * 4);
  bf16* ffp  = (bf16*)take((size_t)MR * FFD * 2);
  // aliases over regions that are dead by the time these are written:
  bf16* ctx = qbuf;  // q bf16 dead after rope
  bf16* h2  = hbuf;  // h dead after qkv gemms
  bf16* ffa = qr;    // spans qr..kr (24MB), dead after attention
  bf16* ffb = qbuf;  // spans qbuf..kbuf (24MB), dead after wo gemm

  auto ct = [&](const float* W, bf16* Wt, int K, int N) {
    size_t tot = (size_t)K * N;
    convtrans_kernel<<<(unsigned)((tot + 255) / 256), 256, 0, stream>>>(W, Wt, K, N);
  };
  ct(wq, wqT, 1024, 2048);
  ct(wk, wkT, 1024, 1024);
  ct(wv, wvT, 1024, 1024);
  ct(wo, woT, 2048, 1024);
  ct(fc1, fc1T, 1024, 3072);
  ct(fc2, fc2T, 1024, 3072);
  ct(fc3, fc3T, 3072, 1024);

  // h = rmsnorm(x) (bf16)
  rmsnorm_kernel<<<MR, 256, 0, stream>>>(x, n1s, hbuf, DD);

  // q/k/v projections  (block tile 128x256)
  gemm_kernel<0><<<dim3(MR / 128, 2048 / 256), 256, 0, stream>>>(hbuf, wqT, MR, 2048, 1024, qbuf, nullptr, nullptr);
  gemm_kernel<0><<<dim3(MR / 128, 1024 / 256), 256, 0, stream>>>(hbuf, wkT, MR, 1024, 1024, kbuf, nullptr, nullptr);
  gemm_kernel<0><<<dim3(MR / 128, 1024 / 256), 256, 0, stream>>>(hbuf, wvT, MR, 1024, 1024, vbuf, nullptr, nullptr);

  // q/k rmsnorm + rope (q pre-scaled by 1/sqrt(HD)); v transpose
  rope_norm_kernel<<<(BB * SS * HH) / 4, 128, 0, stream>>>(qbuf, qr, qns, cosb, sinb, HH, 0.08838834764831845f);
  rope_norm_kernel<<<(BB * SS * KVH) / 4, 128, 0, stream>>>(kbuf, kr, kns, cosb, sinb, KVH, 1.0f);
  vtrans_kernel<<<(MR * KVH * HDim) / 256, 256, 0, stream>>>(vbuf, vtb);

  // flash attention
  attn_kernel<<<dim3(BB * HH, SS / 64), 128, 0, stream>>>(qr, kr, vtb, ctx);

  // x1 = x + ctx @ wo
  gemm_kernel<1><<<dim3(MR / 128, 1024 / 256), 256, 0, stream>>>(ctx, woT, MR, 1024, 2048, nullptr, x1, x);

  // h2 = rmsnorm(x1)
  rmsnorm_kernel<<<MR, 256, 0, stream>>>(x1, n2s, h2, DD);

  // FFN
  gemm_kernel<0><<<dim3(MR / 128, 3072 / 256), 256, 0, stream>>>(h2, fc1T, MR, 3072, 1024, ffa, nullptr, nullptr);
  gemm_kernel<0><<<dim3(MR / 128, 3072 / 256), 256, 0, stream>>>(h2, fc2T, MR, 3072, 1024, ffb, nullptr, nullptr);
  silu_mul_kernel<<<(unsigned)(((size_t)MR * FFD) / 256), 256, 0, stream>>>(ffa, ffb, ffp);

  // out = x1 + ffp @ fc3
  gemm_kernel<1><<<dim3(MR / 128, 1024 / 256), 256, 0, stream>>>(ffp, fc3T, MR, 1024, 3072, nullptr, out, x1);
}
